// get_model_33612414059083
// MI455X (gfx1250) — matrix operand order repaired and checked by reading, not compiled, not run on hardware
//
#include <hip/hip_runtime.h>

// ============================================================================
// PointNet++ MSG forward for MI455X (gfx1250).
// conv_bn_relu / linear layers -> bf16 WMMA GEMMs (f32 accumulate):
//   v_wmma_f32_16x16x32_bf16, one 16x32 output slab per wave32 (2 acc tiles,
//   A fragment reused). A tile (16 x CinPad bf16) is staged into LDS once per
//   block with GLOBAL_LOAD_ASYNC_TO_LDS_B128 (+ s_wait_asynccnt) when the
//   toolchain exposes the gfx1250 builtins, else a plain load/ds-store copy.
// Model is ~30 GFLOP total -> latency/bandwidth bound; wins come from bf16
// operands into the matrix pipes, LDS reuse of A, and fused elementwise.
// ============================================================================

static constexpr int NB = 4;      // batch
static constexpr int N0 = 8192;   // input points

typedef __attribute__((ext_vector_type(16))) __bf16         v16bf;
typedef __attribute__((ext_vector_type(8)))  float          v8f;
typedef __attribute__((ext_vector_type(8)))  unsigned short v8us;
typedef __attribute__((ext_vector_type(16))) unsigned short v16us;
typedef __attribute__((ext_vector_type(4)))  int            v4i_t;

union Frag16 { v16us u; v16bf b; };

#if defined(__has_builtin)
#if __has_builtin(__builtin_amdgcn_global_load_async_to_lds_b128)
#define HAVE_ASYNC_LDS 1
#endif
#if __has_builtin(__builtin_amdgcn_s_wait_asynccnt)
#define HAVE_WAIT_ASYNC 1
#endif
#endif
#ifndef HAVE_ASYNC_LDS
#define HAVE_ASYNC_LDS 0
#endif
#ifndef HAVE_WAIT_ASYNC
#define HAVE_WAIT_ASYNC 0
#endif

#if HAVE_ASYNC_LDS
// Builtin signature (from hipcc diagnostic): arg0 is `int __vector(4)` pointer
// in the global address space (non-const); arg1 is the LDS-side pointer.
typedef __attribute__((address_space(1))) v4i_t* glb_v4i_p;
typedef __attribute__((address_space(3))) v4i_t* lds_v4i_p;

__device__ __forceinline__ void async_copy_b128(const unsigned short* g, unsigned short* l)
{
    __builtin_amdgcn_global_load_async_to_lds_b128(
        (glb_v4i_p)(v4i_t*)(unsigned short*)g,
        (lds_v4i_p)(v4i_t*)l, 0, 0);
}
__device__ __forceinline__ void wait_async_lds()
{
#if HAVE_WAIT_ASYNC
    __builtin_amdgcn_s_wait_asynccnt(0);
#else
    asm volatile("s_wait_asynccnt 0" ::: "memory");
#endif
}
#endif

// ---------------------------------------------------------------------------
// f32 -> bf16 (round-to-nearest-even) with zero padding into [rowsPad,colsPad]
// ---------------------------------------------------------------------------
__global__ __launch_bounds__(256)
void cvt_bf16_pad_kernel(const float* __restrict__ src, unsigned short* __restrict__ dst,
                         int rows, int cols, int rowsPad, int colsPad)
{
    long long i = blockIdx.x * (long long)blockDim.x + threadIdx.x;
    long long total = (long long)rowsPad * colsPad;
    if (i >= total) return;
    int c = (int)(i % colsPad);
    long long r = i / colsPad;
    float v = (r < rows && c < cols) ? src[r * cols + c] : 0.0f;
    unsigned int u = __float_as_uint(v);
    u += 0x7FFFu + ((u >> 16) & 1u);
    dst[i] = (unsigned short)(u >> 16);
}

// ---------------------------------------------------------------------------
// WMMA GEMM: Y[M,Cout] = Abf[M,CinPad] * Wbf[CoutPad,CinPad]^T + bias
// grid = (ceil(npairs/4), tilesM); block = 128 (4 waves).
// Block stages its A tile (16 x CinPad, contiguous in padded A) in LDS via
// async-to-LDS, then each wave computes two adjacent 16x16 N-tiles, reusing
// the A fragment (read from LDS) across both WMMAs.
// A/B fragments use contiguous k = 16*laneHalf + s (same in both); f32 C/D layout per CDNA5 ISA 7.12.2.
// ---------------------------------------------------------------------------
__global__ __launch_bounds__(128)
void wmma_gemm_bias_kernel(const unsigned short* __restrict__ A,
                           const unsigned short* __restrict__ W,
                           const float* __restrict__ bias,
                           float* __restrict__ Y,
                           int M, int Cout, int CinPad, int tilesN, int npairs)
{
    __shared__ unsigned short Atile[16 * 768];   // CinPad <= 768 for this model
    int wave = threadIdx.x >> 5;
    int lane = threadIdx.x & 31;
    int tm = blockIdx.y;

    // ---- cooperative stage of the A tile into LDS (one contiguous region)
    {
        const unsigned short* gsrc = A + (size_t)tm * 16 * CinPad;
        int chunks = (16 * CinPad) >> 3;         // 16-byte chunks
        for (int i = threadIdx.x; i < chunks; i += 128) {
#if HAVE_ASYNC_LDS
            async_copy_b128(gsrc + i * 8, Atile + i * 8);
#else
            *(v8us*)(Atile + i * 8) = *(const v8us*)(gsrc + i * 8);
#endif
        }
#if HAVE_ASYNC_LDS
        wait_async_lds();
#endif
        __syncthreads();
    }

    int pair = blockIdx.x * 4 + wave;            // wave-uniform
    bool active = pair < npairs;
    int cp = active ? pair : (npairs - 1);       // clamp for in-bounds loads
    int tn0 = cp * 2;
    bool has1 = (tn0 + 1) < tilesN;
    int tn1 = has1 ? tn0 + 1 : tn0;

    int hi = (lane >> 4) & 1;                    // lane half selects K sub-block
    int lo = lane & 15;

    const v8us* ArowL = (const v8us*)(Atile + (size_t)lo * CinPad);             // LDS
    const v8us* Wrow0 = (const v8us*)(W + (size_t)(tn0 * 16 + lo) * CinPad);    // global
    const v8us* Wrow1 = (const v8us*)(W + (size_t)(tn1 * 16 + lo) * CinPad);

    v8f acc0 = {}, acc1 = {};
    int kChunks = CinPad >> 5;
    for (int kc = 0; kc < kChunks; ++kc) {
        int base = kc * 4;                       // 4 x v8us per 32 K-elements
        __builtin_prefetch((const void*)(Wrow0 + base + 4), 0, 3);
        __builtin_prefetch((const void*)(Wrow1 + base + 4), 0, 3);
        v8us a0 = ArowL[base + 2 * hi];          // K = k0 + 16*hi     .. +7
        v8us a1 = ArowL[base + 2 * hi + 1];      // K = k0 + 16*hi + 8 .. +7
        v8us b00 = Wrow0[base + 2 * hi];         // K = k0 + 16*hi     .. +7
        v8us b01 = Wrow0[base + 2 * hi + 1];     // K = k0 + 16*hi + 8 .. +7
        v8us b10 = Wrow1[base + 2 * hi];
        v8us b11 = Wrow1[base + 2 * hi + 1];
        Frag16 af, bf0, bf1;
#pragma unroll
        for (int i = 0; i < 8; ++i) {
            af.u[i]  = a0[i];  af.u[i + 8]  = a1[i];
            bf0.u[i] = b00[i]; bf0.u[i + 8] = b01[i];
            bf1.u[i] = b10[i]; bf1.u[i + 8] = b11[i];
        }
        acc0 = __builtin_amdgcn_wmma_f32_16x16x32_bf16(
            false, af.b, false, bf0.b, (short)0, acc0, false, false);
        acc1 = __builtin_amdgcn_wmma_f32_16x16x32_bf16(
            false, af.b, false, bf1.b, (short)0, acc1, false, false);
    }

    if (active) {
        int n0 = tn0 * 16 + lo;
        if (n0 < Cout) {
            float bv = bias ? bias[n0] : 0.0f;
#pragma unroll
            for (int v = 0; v < 8; ++v) {
                int m = tm * 16 + v + hi * 8;    // D layout: M = vgpr + 8*laneHalf
                if (m < M) Y[(size_t)m * Cout + n0] = acc0[v] + bv;
            }
        }
        if (has1) {
            int n1 = tn1 * 16 + lo;
            if (n1 < Cout) {
                float bv = bias ? bias[n1] : 0.0f;
#pragma unroll
                for (int v = 0; v < 8; ++v) {
                    int m = tm * 16 + v + hi * 8;
                    if (m < M) Y[(size_t)m * Cout + n1] = acc1[v] + bv;
                }
            }
        }
    }
}

// ---------------------------------------------------------------------------
// BatchNorm stats: per-channel mean/var over M rows (one block per channel).
// ---------------------------------------------------------------------------
__global__ __launch_bounds__(256)
void bn_stats_kernel(const float* __restrict__ Y, int M, int C,
                     float* __restrict__ mean, float* __restrict__ var)
{
    int c = blockIdx.x;
    __shared__ float ssum[256], ssq[256];
    float s = 0.f, q = 0.f;
    for (int i = threadIdx.x; i < M; i += 256) {
        float v = Y[(size_t)i * C + c];
        s += v; q += v * v;
    }
    ssum[threadIdx.x] = s; ssq[threadIdx.x] = q;
    __syncthreads();
    for (int off = 128; off > 0; off >>= 1) {
        if (threadIdx.x < off) {
            ssum[threadIdx.x] += ssum[threadIdx.x + off];
            ssq[threadIdx.x]  += ssq[threadIdx.x + off];
        }
        __syncthreads();
    }
    if (threadIdx.x == 0) {
        float m = ssum[0] / (float)M;
        mean[c] = m;
        var[c]  = fmaxf(ssq[0] / (float)M - m * m, 0.0f);
    }
}

__global__ __launch_bounds__(256)
void bn_apply_relu_kernel(float* __restrict__ Y, const float* __restrict__ mean,
                          const float* __restrict__ var, const float* __restrict__ g,
                          const float* __restrict__ be, long long total, int C)
{
    long long i = blockIdx.x * (long long)blockDim.x + threadIdx.x;
    if (i >= total) return;
    int c = (int)(i % C);
    float y = Y[i];
    y = g[c] * (y - mean[c]) * rsqrtf(var[c] + 1e-5f) + be[c];
    Y[i] = fmaxf(y, 0.0f);
}

// ---------------------------------------------------------------------------
// Farthest point sampling (matches lax.scan semantics: first index = 0,
// outputs the index *before* each update; argmax ties -> lowest index).
// One block per batch.
// ---------------------------------------------------------------------------
__global__ __launch_bounds__(256)
void fps_kernel(const float* __restrict__ xyz, int N, int npoint, int* __restrict__ out)
{
    int b = blockIdx.x;
    const float* X = xyz + (size_t)b * N * 3;
    __shared__ float sval[256];
    __shared__ int   sidx[256];
    float dist[32];
    int items = (N + 255) / 256;
    for (int j = 0; j < items; ++j) dist[j] = 1e10f;
    int far = 0;
    for (int it = 0; it < npoint; ++it) {
        if (threadIdx.x == 0) out[b * npoint + it] = far;
        float cx = X[far * 3 + 0], cy = X[far * 3 + 1], cz = X[far * 3 + 2];
        float best = -1.0f; int bi = 0;
        for (int j = 0; j < items; ++j) {
            int i = threadIdx.x + j * 256;
            if (i < N) {
                float dx = X[i * 3] - cx, dy = X[i * 3 + 1] - cy, dz = X[i * 3 + 2] - cz;
                float d = dx * dx + dy * dy + dz * dz;
                float dm = fminf(dist[j], d);
                dist[j] = dm;
                if (dm > best) { best = dm; bi = i; }
            }
        }
        sval[threadIdx.x] = best; sidx[threadIdx.x] = bi;
        __syncthreads();
        for (int off = 128; off > 0; off >>= 1) {
            if (threadIdx.x < off) {
                float v2 = sval[threadIdx.x + off]; int i2 = sidx[threadIdx.x + off];
                if (v2 > sval[threadIdx.x] ||
                    (v2 == sval[threadIdx.x] && i2 < sidx[threadIdx.x])) {
                    sval[threadIdx.x] = v2; sidx[threadIdx.x] = i2;
                }
            }
            __syncthreads();
        }
        far = sidx[0];
        __syncthreads();
    }
}

__global__ __launch_bounds__(256)
void gather_centers_kernel(const float* __restrict__ xyz, const int* __restrict__ fidx,
                           float* __restrict__ nxyz, int N, int S, int total)
{
    int t = blockIdx.x * blockDim.x + threadIdx.x;   // over B*S
    if (t >= total) return;
    int b = t / S;
    int i = fidx[t];
    const float* p = xyz + ((size_t)b * N + i) * 3;
    nxyz[(size_t)t * 3 + 0] = p[0];
    nxyz[(size_t)t * 3 + 1] = p[1];
    nxyz[(size_t)t * 3 + 2] = p[2];
}

// ---------------------------------------------------------------------------
// Ball query: first K indices (ascending) with d^2 <= r^2; pad with first hit.
// ---------------------------------------------------------------------------
__global__ __launch_bounds__(256)
void query_ball_kernel(const float* __restrict__ xyz, const float* __restrict__ nxyz,
                       int N, int S, float r2, int K, int* __restrict__ idx, int total)
{
    int t = blockIdx.x * blockDim.x + threadIdx.x;   // over B*S
    if (t >= total) return;
    int b = t / S;
    const float* X = xyz + (size_t)b * N * 3;
    float cx = nxyz[(size_t)t * 3 + 0], cy = nxyz[(size_t)t * 3 + 1], cz = nxyz[(size_t)t * 3 + 2];
    int cnt = 0, first = 0;
    for (int i = 0; i < N && cnt < K; ++i) {
        float dx = X[i * 3] - cx, dy = X[i * 3 + 1] - cy, dz = X[i * 3 + 2] - cz;
        float d = dx * dx + dy * dy + dz * dz;
        if (d <= r2) {
            if (cnt == 0) first = i;
            idx[(size_t)t * K + cnt] = i;
            ++cnt;
        }
    }
    for (int k = cnt; k < K; ++k) idx[(size_t)t * K + k] = first;
}

// g[b,s,k,:] = concat(points[idx], xyz[idx] - center)
__global__ __launch_bounds__(256)
void group_kernel(const float* __restrict__ xyz, const float* __restrict__ pts,
                  const float* __restrict__ nxyz, const int* __restrict__ idx,
                  float* __restrict__ g, int N, int S, int K, int Cpts, long long total)
{
    long long t = blockIdx.x * (long long)blockDim.x + threadIdx.x;  // over B*S*K
    if (t >= total) return;
    long long bs = t / K;
    int b = (int)(bs / S);
    int i = idx[t];
    const float* ps = pts + ((size_t)b * N + i) * Cpts;
    float* gr = g + (size_t)t * (Cpts + 3);
    for (int c = 0; c < Cpts; ++c) gr[c] = ps[c];
    const float* xs = xyz + ((size_t)b * N + i) * 3;
    const float* ct = nxyz + (size_t)bs * 3;
    gr[Cpts + 0] = xs[0] - ct[0];
    gr[Cpts + 1] = xs[1] - ct[1];
    gr[Cpts + 2] = xs[2] - ct[2];
}

// max over K neighbors, writing into concat slot [coff, coff+C) of out row
__global__ __launch_bounds__(256)
void maxpool_k_kernel(const float* __restrict__ in, float* __restrict__ out,
                      int K, int C, int Ctot, int coff, long long total)
{
    long long i = blockIdx.x * (long long)blockDim.x + threadIdx.x;  // over P*C
    if (i >= total) return;
    long long p = i / C; int c = (int)(i % C);
    float m = -3.4e38f;
    for (int k = 0; k < K; ++k) m = fmaxf(m, in[((size_t)p * K + k) * C + c]);
    out[(size_t)p * Ctot + coff + c] = m;
}

__global__ __launch_bounds__(256)
void maxpool_all_kernel(const float* __restrict__ in, float* __restrict__ out,
                        int S, int C, int total)
{
    int i = blockIdx.x * blockDim.x + threadIdx.x;   // over B*C
    if (i >= total) return;
    int b = i / C, c = i % C;
    float m = -3.4e38f;
    for (int s = 0; s < S; ++s) m = fmaxf(m, in[((size_t)b * S + s) * C + c]);
    out[i] = m;
}

__global__ __launch_bounds__(256)
void copy_concat_kernel(const float* __restrict__ src, float* __restrict__ dst,
                        int Csrc, int Ctot, int coff, long long total)
{
    long long i = blockIdx.x * (long long)blockDim.x + threadIdx.x;  // rows*Csrc
    if (i >= total) return;
    long long r = i / Csrc; int c = (int)(i % Csrc);
    dst[r * Ctot + coff + c] = src[i];
}

// 3-NN inverse-distance interpolation (reference top_k tie behavior: strict <)
__global__ __launch_bounds__(256)
void three_interp_kernel(const float* __restrict__ xyz1, const float* __restrict__ xyz2,
                         const float* __restrict__ pts2, float* __restrict__ out,
                         int S1, int S2, int C2, int Ctot, int coff, long long total)
{
    long long t = blockIdx.x * (long long)blockDim.x + threadIdx.x;  // over B*S1
    if (t >= total) return;
    int b = (int)(t / S1);
    const float* p1 = xyz1 + (size_t)t * 3;
    const float* X2 = xyz2 + (size_t)b * S2 * 3;
    float cx = p1[0], cy = p1[1], cz = p1[2];
    float d0 = 1e30f, d1 = 1e30f, d2 = 1e30f;
    int i0 = 0, i1 = 0, i2 = 0;
    for (int i = 0; i < S2; ++i) {
        float dx = X2[i * 3] - cx, dy = X2[i * 3 + 1] - cy, dz = X2[i * 3 + 2] - cz;
        float d = fmaxf(dx * dx + dy * dy + dz * dz, 0.0f);
        if (d < d0)      { d2 = d1; i2 = i1; d1 = d0; i1 = i0; d0 = d; i0 = i; }
        else if (d < d1) { d2 = d1; i2 = i1; d1 = d;  i1 = i; }
        else if (d < d2) { d2 = d;  i2 = i; }
    }
    float w0 = 1.0f / (d0 + 1e-8f), w1 = 1.0f / (d1 + 1e-8f), w2 = 1.0f / (d2 + 1e-8f);
    float ws = w0 + w1 + w2; w0 /= ws; w1 /= ws; w2 /= ws;
    const float* f0 = pts2 + ((size_t)b * S2 + i0) * C2;
    const float* f1 = pts2 + ((size_t)b * S2 + i1) * C2;
    const float* f2 = pts2 + ((size_t)b * S2 + i2) * C2;
    float* orow = out + (size_t)t * Ctot + coff;
    for (int c = 0; c < C2; ++c) orow[c] = w0 * f0[c] + w1 * f1[c] + w2 * f2[c];
}

// per-row LayerNorm + ReLU, C == blockDim.x == 256
__global__ __launch_bounds__(256)
void layernorm_relu_kernel(float* __restrict__ x, const float* __restrict__ g,
                           const float* __restrict__ be, int C)
{
    __shared__ float red[256];
    int b = blockIdx.x;
    float v = x[b * C + threadIdx.x];
    red[threadIdx.x] = v; __syncthreads();
    for (int off = 128; off > 0; off >>= 1) {
        if (threadIdx.x < off) red[threadIdx.x] += red[threadIdx.x + off];
        __syncthreads();
    }
    float m = red[0] / (float)C; __syncthreads();
    float d = v - m;
    red[threadIdx.x] = d * d; __syncthreads();
    for (int off = 128; off > 0; off >>= 1) {
        if (threadIdx.x < off) red[threadIdx.x] += red[threadIdx.x + off];
        __syncthreads();
    }
    float var = red[0] / (float)C;
    float y = g[threadIdx.x] * d * rsqrtf(var + 1e-5f) + be[threadIdx.x];
    x[b * C + threadIdx.x] = fmaxf(y, 0.0f);
}

// [B,S,C] -> [B,C,S]
__global__ __launch_bounds__(256)
void transpose_out_kernel(const float* __restrict__ in, float* __restrict__ out,
                          int S, int C, long long total)
{
    long long i = blockIdx.x * (long long)blockDim.x + threadIdx.x;
    if (i >= total) return;
    long long sc = (long long)S * C;
    int b = (int)(i / sc);
    long long r = i % sc;
    int s = (int)(r / C); int c = (int)(r % C);
    out[(size_t)b * sc + (size_t)c * S + s] = in[i];
}

// input [B,6,N] -> point-major l0_xyz [B,N,3], l0_pts [B,N,6]
__global__ __launch_bounds__(256)
void build_l0_kernel(const float* __restrict__ in, float* __restrict__ xyz_pm,
                     float* __restrict__ pts_pm, int N, long long total)
{
    long long i = blockIdx.x * (long long)blockDim.x + threadIdx.x;  // B*N
    if (i >= total) return;
    int b = (int)(i / N); int n = (int)(i % N);
    const float* src = in + (size_t)b * 6 * N + n;
    float* pd = pts_pm + i * 6;
#pragma unroll
    for (int c = 0; c < 6; ++c) pd[c] = src[(size_t)c * N];
    float* xd = xyz_pm + i * 3;
    xd[0] = pd[0]; xd[1] = pd[1]; xd[2] = pd[2];
}

// out2 = first 3 channels of input (already [B,6,N] channel-major)
__global__ __launch_bounds__(256)
void copy_xyz_out_kernel(const float* __restrict__ in, float* __restrict__ out,
                         int N, long long total)
{
    long long i = blockIdx.x * (long long)blockDim.x + threadIdx.x;  // B*3*N
    if (i >= total) return;
    long long per = 3LL * N;
    int b = (int)(i / per);
    long long r = i % per;
    out[i] = in[(size_t)b * 6 * N + r];
}

__global__ __launch_bounds__(256)
void copy_kernel(const float* __restrict__ src, float* __restrict__ dst, long long n)
{
    long long i = blockIdx.x * (long long)blockDim.x + threadIdx.x;
    if (i < n) dst[i] = src[i];
}

// ============================================================================
// Host-side orchestration
// ============================================================================
struct CB { const float *b, *be, *g, *w; int cin, cout; };
struct Scratch { unsigned short *xbf, *wbf; float *mean, *var; };

static inline int gblk(long long n) { return (int)((n + 255) / 256); }

// GEMM (+ optional BN+ReLU): in [M,cin] f32 -> out [M,cout] f32
static void conv_layer(hipStream_t st, const Scratch& sc, const float* in, float* out,
                       const CB& p, int M, bool bn_relu)
{
    int CinPad  = (p.cin + 31) & ~31;
    int CoutPad = (p.cout + 15) & ~15;
    int Mpad    = (M + 15) & ~15;
    long long tx = (long long)Mpad * CinPad;
    cvt_bf16_pad_kernel<<<gblk(tx), 256, 0, st>>>(in, sc.xbf, M, p.cin, Mpad, CinPad);
    long long tw = (long long)CoutPad * CinPad;
    cvt_bf16_pad_kernel<<<gblk(tw), 256, 0, st>>>(p.w, sc.wbf, p.cout, p.cin, CoutPad, CinPad);
    int tilesM = Mpad / 16, tilesN = CoutPad / 16;
    int npairs = (tilesN + 1) / 2;
    dim3 grid((npairs + 3) / 4, tilesM);
    wmma_gemm_bias_kernel<<<grid, 128, 0, st>>>(sc.xbf, sc.wbf, p.b, out,
                                                M, p.cout, CinPad, tilesN, npairs);
    if (bn_relu) {
        bn_stats_kernel<<<p.cout, 256, 0, st>>>(out, M, p.cout, sc.mean, sc.var);
        long long ty = (long long)M * p.cout;
        bn_apply_relu_kernel<<<gblk(ty), 256, 0, st>>>(out, sc.mean, sc.var, p.g, p.be, ty, p.cout);
    }
}

static void sa_stage(hipStream_t st, const Scratch& sc,
                     const float* xyz_in, const float* pts_in, int Nin, int Cpts, int Snew,
                     const float* radii, const int* Ks, const CB (*br)[3],
                     int* fidx, int* ballidx, float* gbuf, float* Y0, float* Y1,
                     float* out_xyz, float* out_pts)
{
    fps_kernel<<<NB, 256, 0, st>>>(xyz_in, Nin, Snew, fidx);
    int BS = NB * Snew;
    gather_centers_kernel<<<gblk(BS), 256, 0, st>>>(xyz_in, fidx, out_xyz, Nin, Snew, BS);
    int Ctot = br[0][2].cout + br[1][2].cout;
    int coff = 0;
    for (int j = 0; j < 2; ++j) {
        int K = Ks[j];
        float r2 = radii[j] * radii[j];
        query_ball_kernel<<<gblk(BS), 256, 0, st>>>(xyz_in, out_xyz, Nin, Snew, r2, K, ballidx, BS);
        long long M = (long long)BS * K;
        group_kernel<<<gblk(M), 256, 0, st>>>(xyz_in, pts_in, out_xyz, ballidx, gbuf,
                                              Nin, Snew, K, Cpts, M);
        const float* cur = gbuf;
        float* dsts[3] = {Y0, Y1, Y0};
        for (int L = 0; L < 3; ++L) {
            conv_layer(st, sc, cur, dsts[L], br[j][L], (int)M, true);
            cur = dsts[L];
        }
        int lastC = br[j][2].cout;
        long long tp = (long long)BS * lastC;
        maxpool_k_kernel<<<gblk(tp), 256, 0, st>>>(cur, out_pts, K, lastC, Ctot, coff, tp);
        coff += lastC;
    }
}

static void fp_stage(hipStream_t st, const Scratch& sc,
                     const float* xyz1, const float* xyz2,
                     const float* pts1, const float* pts2,
                     int S1, int S2, int C1, int C2, const CB* layers,
                     float* gbuf, float* Y0, float* Y1, float* out)
{
    int Ctot = C1 + C2;
    long long rows = (long long)NB * S1;
    long long t1 = rows * C1;
    copy_concat_kernel<<<gblk(t1), 256, 0, st>>>(pts1, gbuf, C1, Ctot, 0, t1);
    three_interp_kernel<<<gblk(rows), 256, 0, st>>>(xyz1, xyz2, pts2, gbuf,
                                                    S1, S2, C2, Ctot, C1, rows);
    const float* cur = gbuf;
    float* dsts[3] = {Y0, Y1, out};
    for (int L = 0; L < 3; ++L) {
        conv_layer(st, sc, cur, dsts[L], layers[L], (int)rows, true);
        cur = dsts[L];
    }
}

extern "C" void kernel_launch(void* const* d_in, const int* in_sizes, int n_in,
                              void* d_out, int out_size, void* d_ws, size_t ws_size,
                              hipStream_t stream)
{
    (void)in_sizes; (void)n_in; (void)out_size; (void)ws_size;

    // -------- unpack inputs (jax pytree order: sorted dict keys, lists in order)
    const float* xyz_in = (const float*)d_in[0];
    int t = 1;
    auto nxt   = [&]() { return (const float*)d_in[t++]; };
    auto getCB = [&](int ci, int co) {
        CB c; c.b = nxt(); c.be = nxt(); c.g = nxt(); c.w = nxt(); c.cin = ci; c.cout = co; return c;
    };
    CB dist1 = getCB(512, 512);
    CB dist2 = getCB(512, 256);
    CB dist_res; dist_res.b = nxt(); dist_res.w = nxt();
    dist_res.g = dist_res.be = nullptr; dist_res.cin = 256; dist_res.cout = 16;
    CB fc1; fc1.b = nxt(); fc1.w = nxt(); fc1.g = fc1.be = nullptr; fc1.cin = 512; fc1.cout = 256;
    CB fc2; fc2.b = nxt(); fc2.w = nxt(); fc2.g = fc2.be = nullptr; fc2.cin = 256; fc2.cout = 16;
    CB fp1L[3] = { getCB(134, 134), getCB(134, 64), getCB(64, 32) };
    CB fp2L[3] = { getCB(320, 320), getCB(320, 128), getCB(128, 128) };
    CB fp3L[3] = { getCB(768, 768), getCB(768, 256), getCB(256, 256) };
    CB fsaL[2] = { getCB(515, 256), getCB(256, 512) };
    const float* ln_be = nxt();
    const float* ln_g  = nxt();
    CB off1 = getCB(512, 512);
    CB off2 = getCB(512, 256);
    CB off_res; off_res.b = nxt(); off_res.w = nxt();
    off_res.g = off_res.be = nullptr; off_res.cin = 256; off_res.cout = 48;
    const int sa1d[4] = {9, 6, 32, 32};
    const int sa2d[4] = {67, 64, 64, 128};
    const int sa3d[4] = {259, 256, 196, 256};
    CB sa1b[2][3], sa2b[2][3], sa3b[2][3];
    for (int j = 0; j < 2; ++j) for (int L = 0; L < 3; ++L) sa1b[j][L] = getCB(sa1d[L], sa1d[L + 1]);
    for (int j = 0; j < 2; ++j) for (int L = 0; L < 3; ++L) sa2b[j][L] = getCB(sa2d[L], sa2d[L + 1]);
    for (int j = 0; j < 2; ++j) for (int L = 0; L < 3; ++L) sa3b[j][L] = getCB(sa3d[L], sa3d[L + 1]);

    // -------- workspace bump allocator
    char* wp = (char*)d_ws;
    auto alloc = [&](size_t bytes) -> void* {
        void* p = wp; wp += (bytes + 255) & ~(size_t)255; return p;
    };
    float* l0_xyz = (float*)alloc((size_t)NB * N0 * 3 * 4);
    float* l0_pts = (float*)alloc((size_t)NB * N0 * 6 * 4);
    float* l1_xyz = (float*)alloc((size_t)NB * 1024 * 3 * 4);
    float* l1_pts = (float*)alloc((size_t)NB * 1024 * 64 * 4);
    float* l2_xyz = (float*)alloc((size_t)NB * 512 * 3 * 4);
    float* l2_pts = (float*)alloc((size_t)NB * 512 * 256 * 4);
    float* l3_xyz = (float*)alloc((size_t)NB * 256 * 3 * 4);
    float* l3_pts = (float*)alloc((size_t)NB * 256 * 512 * 4);
    float* l2_up  = (float*)alloc((size_t)NB * 512 * 256 * 4);
    float* l1_up  = (float*)alloc((size_t)NB * 1024 * 128 * 4);
    float* l0_out = (float*)alloc((size_t)NB * N0 * 32 * 4);
    float* off_out = (float*)alloc((size_t)NB * 256 * 48 * 4);
    float* dst_out = (float*)alloc((size_t)NB * 256 * 16 * 4);
    float* ep_pool = (float*)alloc((size_t)NB * 512 * 4);
    float* ep1     = (float*)alloc((size_t)NB * 256 * 4);
    float* ep2     = (float*)alloc((size_t)NB * 16 * 4);
    int* fidx    = (int*)alloc((size_t)NB * 1024 * 4);
    int* ballidx = (int*)alloc((size_t)NB * 1024 * 32 * 4);
    float* gbuf = (float*)alloc((size_t)9000000 * 4);   // max 32768x259
    float* Y0   = (float*)alloc((size_t)8500000 * 4);   // max 65536x128 / 32768x256
    float* Y1   = (float*)alloc((size_t)8500000 * 4);
    unsigned short* xbf = (unsigned short*)alloc((size_t)9500000 * 2); // max 32768x288
    unsigned short* wbf = (unsigned short*)alloc((size_t)600000 * 2);  // max 768x768
    float* meanb = (float*)alloc(768 * 4);
    float* varb  = (float*)alloc(768 * 4);
    Scratch sc{xbf, wbf, meanb, varb};

    // -------- l0: transpose input into point-major layout
    {
        long long tn = (long long)NB * N0;
        build_l0_kernel<<<gblk(tn), 256, 0, stream>>>(xyz_in, l0_xyz, l0_pts, N0, tn);
    }

    // -------- SA stages (MSG)
    const float r_sa1[2] = {0.0025f, 0.005f};
    const float r_sa2[2] = {0.005f, 0.01f};
    const float r_sa3[2] = {0.01f, 0.02f};
    const int   Ks[2]    = {16, 32};
    sa_stage(stream, sc, l0_xyz, l0_pts, N0, 6, 1024, r_sa1, Ks, sa1b,
             fidx, ballidx, gbuf, Y0, Y1, l1_xyz, l1_pts);
    sa_stage(stream, sc, l1_xyz, l1_pts, 1024, 64, 512, r_sa2, Ks, sa2b,
             fidx, ballidx, gbuf, Y0, Y1, l2_xyz, l2_pts);
    sa_stage(stream, sc, l2_xyz, l2_pts, 512, 256, 256, r_sa3, Ks, sa3b,
             fidx, ballidx, gbuf, Y0, Y1, l3_xyz, l3_pts);

    // -------- FP stages
    fp_stage(stream, sc, l2_xyz, l3_xyz, l2_pts, l3_pts, 512, 256, 256, 512, fp3L,
             gbuf, Y0, Y1, l2_up);
    fp_stage(stream, sc, l1_xyz, l2_xyz, l1_pts, l2_up, 1024, 512, 64, 256, fp2L,
             gbuf, Y0, Y1, l1_up);
    fp_stage(stream, sc, l0_xyz, l1_xyz, l0_pts, l1_up, N0, 1024, 6, 128, fp1L,
             gbuf, Y0, Y1, l0_out);

    // -------- heads on l3_pts [B*256, 512]
    int Mh = NB * 256;
    conv_layer(stream, sc, l3_pts, Y0, off1, Mh, true);
    conv_layer(stream, sc, Y0, Y1, off2, Mh, true);
    conv_layer(stream, sc, Y1, off_out, off_res, Mh, false);
    conv_layer(stream, sc, l3_pts, Y0, dist1, Mh, true);
    conv_layer(stream, sc, Y0, Y1, dist2, Mh, true);
    conv_layer(stream, sc, Y1, dst_out, dist_res, Mh, false);

    // -------- global SA + FC head
    {
        long long rows = (long long)NB * 256;
        long long ta = rows * 3;
        copy_concat_kernel<<<gblk(ta), 256, 0, stream>>>(l3_xyz, gbuf, 3, 515, 0, ta);
        long long tb = rows * 512;
        copy_concat_kernel<<<gblk(tb), 256, 0, stream>>>(l3_pts, gbuf, 512, 515, 3, tb);
        conv_layer(stream, sc, gbuf, Y0, fsaL[0], (int)rows, true);
        conv_layer(stream, sc, Y0, Y1, fsaL[1], (int)rows, true);
        maxpool_all_kernel<<<gblk(NB * 512), 256, 0, stream>>>(Y1, ep_pool, 256, 512, NB * 512);
        conv_layer(stream, sc, ep_pool, ep1, fc1, NB, false);   // M=4, padded to 16 internally
        layernorm_relu_kernel<<<NB, 256, 0, stream>>>(ep1, ln_g, ln_be, 256);
        conv_layer(stream, sc, ep1, ep2, fc2, NB, false);
    }

    // -------- assemble outputs (tuple order, channel-major transposes)
    float* out = (float*)d_out;
    size_t o = 0;
    { long long n = (long long)NB * N0 * 32;
      transpose_out_kernel<<<gblk(n), 256, 0, stream>>>(l0_out, out + o, N0, 32, n);  o += n; }
    { long long n = (long long)NB * 256 * 512;
      transpose_out_kernel<<<gblk(n), 256, 0, stream>>>(l3_pts, out + o, 256, 512, n); o += n; }
    { long long n = (long long)NB * 3 * N0;
      copy_xyz_out_kernel<<<gblk(n), 256, 0, stream>>>(xyz_in, out + o, N0, n);        o += n; }
    { long long n = (long long)NB * 256 * 3;
      transpose_out_kernel<<<gblk(n), 256, 0, stream>>>(l3_xyz, out + o, 256, 3, n);   o += n; }
    { long long n = (long long)NB * 256 * 48;
      transpose_out_kernel<<<gblk(n), 256, 0, stream>>>(off_out, out + o, 256, 48, n); o += n; }
    { long long n = (long long)NB * 256 * 16;
      transpose_out_kernel<<<gblk(n), 256, 0, stream>>>(dst_out, out + o, 256, 16, n); o += n; }
    { long long n = (long long)NB * 16;
      copy_kernel<<<1, 256, 0, stream>>>(ep2, out + o, n); }
}
